// MultiHeadLatentAttention_21242908246647
// MI455X (gfx1250) — compile-verified, matrix operand order checked by reading, not run on hardware
//
#include <hip/hip_runtime.h>
#include <hip/hip_bf16.h>

// ---------------- types ----------------
typedef _Float16 half16 __attribute__((ext_vector_type(16)));
typedef _Float16 half8  __attribute__((ext_vector_type(8)));
typedef float    f32x8  __attribute__((ext_vector_type(8)));

#define D_MODEL 1024
#define LSEQ    2048
#define BATCH   2
#define NROWS   (BATCH * LSEQ)      // 4096
#define QPROJ   512
#define KVP     682
#define KVP_PAD 704
#define NHEADS  16
#define DH      64

// -------------------------------------------------------------------------
// 16x32 f16 fragment load (A-style layout, CDNA5 WMMA):
//   lane L: row = L&15, half = L>>4
//   elements 0..7  = K[half*8 .. half*8+7]
//   elements 8..15 = K[16+half*8 .. 16+half*8+7]
// base points at (row0, k0) of the tile; ld = row stride in elements.
// All call sites guarantee 16B alignment.
// -------------------------------------------------------------------------
static __device__ __forceinline__ half16 load_frag(const _Float16* __restrict__ base, int ld) {
    const int lane = threadIdx.x & 31;
    const int r = lane & 15;
    const int hf = lane >> 4;
    const _Float16* p = base + (size_t)r * ld + hf * 8;
    half16 f;
    half8* fp = (half8*)&f;
    fp[0] = *(const half8*)(p);
    fp[1] = *(const half8*)(p + 16);
    return f;
}

static __device__ __forceinline__ f32x8 wmma16(half16 a, half16 b, f32x8 c) {
    // D = A(16x32 f16) x B(32x16 f16) + C(16x16 f32)
    return __builtin_amdgcn_wmma_f32_16x16x32_f16(false, a, false, b, (short)0, c, false, false);
}

// ---------------- prep: x f32 -> (out copy, f16) ----------------
__global__ __launch_bounds__(256) void convert_x_kernel(const float* __restrict__ x,
                                                        float* __restrict__ out_x,
                                                        _Float16* __restrict__ x_h, int n) {
    int i = blockIdx.x * 256 + threadIdx.x;
    if (i < n) {
        float v = x[i];
        out_x[i] = v;
        x_h[i]   = (_Float16)v;
    }
}

// ---------------- prep: transpose weight f32[K x N] -> f16[Nout x Kpad] ----
// out[n][kp] = (kp<Kin && n<Nvalid) ? in[kp*ldin + n] : 0
__global__ void transpose_w_kernel(const float* __restrict__ in, _Float16* __restrict__ out,
                                   int Nout, int Kin, int ldin, int Nvalid, int Kpad) {
    int kp = blockIdx.x * 16 + threadIdx.x;
    int n  = blockIdx.y * 16 + threadIdx.y;
    if (kp < Kpad && n < Nout) {
        float v = (kp < Kin && n < Nvalid) ? in[(size_t)kp * ldin + n] : 0.0f;
        out[(size_t)n * Kpad + kp] = (_Float16)v;
    }
}

// ---------------- WMMA GEMM: C[MxN] = A[MxK] * Bt[NxK]^T ----------------
// A,Bt f16 row-major with stride K. 256 threads = 8 waves in 4(M) x 2(N),
// block tile 128x64, wave tile 32x32 (4 accumulators -> 16 flops/byte of
// fragment traffic), K step 32.
__global__ __launch_bounds__(256) void gemm_wmma_kernel(const _Float16* __restrict__ A,
                                                        const _Float16* __restrict__ Bt,
                                                        void* __restrict__ Cout,
                                                        int K, int ldc, int out_f16) {
    const int w  = threadIdx.x >> 5;
    const int wm = w & 3, wn = w >> 2;
    const int m0 = blockIdx.y * 128 + wm * 32;
    const int n0 = blockIdx.x * 64 + wn * 32;

    f32x8 acc00 = {0.f,0.f,0.f,0.f,0.f,0.f,0.f,0.f};
    f32x8 acc01 = acc00, acc10 = acc00, acc11 = acc00;
    const _Float16* aRow0 = A  + (size_t)m0 * K;
    const _Float16* aRow1 = A  + (size_t)(m0 + 16) * K;
    const _Float16* bRow0 = Bt + (size_t)n0 * K;
    const _Float16* bRow1 = Bt + (size_t)(n0 + 16) * K;
    for (int k0 = 0; k0 < K; k0 += 32) {
        half16 a0 = load_frag(aRow0 + k0, K);
        half16 a1 = load_frag(aRow1 + k0, K);
        half16 b0 = load_frag(bRow0 + k0, K);
        half16 b1 = load_frag(bRow1 + k0, K);
        acc00 = wmma16(a0, b0, acc00);
        acc01 = wmma16(a0, b1, acc01);
        acc10 = wmma16(a1, b0, acc10);
        acc11 = wmma16(a1, b1, acc11);
    }

    const int lane = threadIdx.x & 31;
    const int r = lane & 15, hf = lane >> 4;
    f32x8 accs[2][2] = {{acc00, acc01}, {acc10, acc11}};
    if (out_f16) {
        _Float16* C = (_Float16*)Cout;
        #pragma unroll
        for (int mi = 0; mi < 2; mi++)
            #pragma unroll
            for (int ni = 0; ni < 2; ni++)
                #pragma unroll
                for (int v = 0; v < 8; v++)
                    C[(size_t)(m0 + mi * 16 + v + 8 * hf) * ldc + n0 + ni * 16 + r] =
                        (_Float16)accs[mi][ni][v];
    } else {
        float* C = (float*)Cout;
        #pragma unroll
        for (int mi = 0; mi < 2; mi++)
            #pragma unroll
            for (int ni = 0; ni < 2; ni++)
                #pragma unroll
                for (int v = 0; v < 8; v++)
                    C[(size_t)(m0 + mi * 16 + v + 8 * hf) * ldc + n0 + ni * 16 + r] =
                        accs[mi][ni][v];
    }
}

// ---------------- LayerNorm over one row ----------------
__global__ __launch_bounds__(256) void ln_kernel(const float* __restrict__ in, int ld_in, int width,
                                                 const float* __restrict__ gamma,
                                                 const float* __restrict__ beta,
                                                 float* __restrict__ out32, int ld32,
                                                 _Float16* __restrict__ out16, int ld16, int padw) {
    const int row = blockIdx.x;
    const int tid = threadIdx.x;
    const float* xr = in + (size_t)row * ld_in;
    __shared__ float s1[256], s2[256];
    float a = 0.f, b = 0.f;
    for (int c = tid; c < width; c += 256) {
        float v = xr[c];
        a += v; b += v * v;
    }
    s1[tid] = a; s2[tid] = b;
    __syncthreads();
    for (int s = 128; s > 0; s >>= 1) {
        if (tid < s) { s1[tid] += s1[tid + s]; s2[tid] += s2[tid + s]; }
        __syncthreads();
    }
    const float mean = s1[0] / (float)width;
    const float var  = s2[0] / (float)width - mean * mean;
    const float rstd = rsqrtf(var + 1e-5f);
    for (int c = tid; c < padw; c += 256) {
        float o = 0.f;
        if (c < width) {
            o = (xr[c] - mean) * rstd * gamma[c] + beta[c];
            if (out32) out32[(size_t)row * ld32 + c] = o;
        }
        out16[(size_t)row * ld16 + c] = (_Float16)o;
    }
}

// ---------------- fused scores + softmax ----------------
// grid (64 strips, 16 heads, 2 batch), 256 threads = 8 waves.
// Each block: 32 rows x 2048 cols of scores for (b,h). Each wave owns a
// 256-col range and computes BOTH 16-row m-tiles, so every K-fragment load is
// reused twice. Scores go to a ~263KB LDS strip (CDNA5: up to 320KB/WG),
// then row softmax and a single coalesced write of attn.
__global__ __launch_bounds__(256) void attn_kernel(const _Float16* __restrict__ Qh,
                                                   const _Float16* __restrict__ Kh,
                                                   float* __restrict__ attn_out) {
    const int b = blockIdx.z, h = blockIdx.y, strip = blockIdx.x;
    const int m0 = strip * 32;
    const int tid = threadIdx.x;
    const int wave = tid >> 5, lane = tid & 31;
    const int r = lane & 15, hf = lane >> 4;

    __shared__ float strip_s[32][2052];   // +4 pad vs 2048 to break bank stride
    __shared__ float red[32][8];
    __shared__ float rowinv[32];

    // A fragments: Q rows m0..m0+31 (two 16-row m-tiles), head slice, DH=64
    const _Float16* Qbase = Qh + ((size_t)(b * LSEQ + m0)) * D_MODEL + h * DH;
    half16 a00 = load_frag(Qbase, D_MODEL);
    half16 a01 = load_frag(Qbase + 32, D_MODEL);
    half16 a10 = load_frag(Qbase + (size_t)16 * D_MODEL, D_MODEL);
    half16 a11 = load_frag(Qbase + (size_t)16 * D_MODEL + 32, D_MODEL);

    const int n_base = wave * 256;
    #pragma unroll 4
    for (int t = 0; t < 16; t++) {
        const int n0 = n_base + t * 16;
        const _Float16* Kbase = Kh + ((size_t)(b * LSEQ + n0)) * D_MODEL + h * DH;
        half16 b0 = load_frag(Kbase, D_MODEL);
        half16 b1 = load_frag(Kbase + 32, D_MODEL);
        f32x8 acc0 = {0.f,0.f,0.f,0.f,0.f,0.f,0.f,0.f};
        f32x8 acc1 = acc0;
        acc0 = wmma16(a00, b0, acc0);
        acc0 = wmma16(a01, b1, acc0);
        acc1 = wmma16(a10, b0, acc1);
        acc1 = wmma16(a11, b1, acc1);
        #pragma unroll
        for (int v = 0; v < 8; v++) {
            strip_s[v + 8 * hf][n0 + r]      = acc0[v] * 0.125f;  // 1/sqrt(64)
            strip_s[16 + v + 8 * hf][n0 + r] = acc1[v] * 0.125f;
        }
    }
    __syncthreads();

    // softmax: 8 threads per row (32 rows), each covers 256 strided cols
    const int row = tid >> 3, sub = tid & 7;
    float mx = -3.402823466e+38f;
    for (int c = sub; c < LSEQ; c += 8) mx = fmaxf(mx, strip_s[row][c]);
    red[row][sub] = mx;
    __syncthreads();
    float m = red[row][0];
    #pragma unroll
    for (int i = 1; i < 8; i++) m = fmaxf(m, red[row][i]);
    float sm = 0.f;
    for (int c = sub; c < LSEQ; c += 8) {
        float e = __expf(strip_s[row][c] - m);
        strip_s[row][c] = e;
        sm += e;
    }
    __syncthreads();            // done reading red before overwrite
    red[row][sub] = sm;
    __syncthreads();
    if (sub == 0) {
        float s = red[row][0];
        #pragma unroll
        for (int i = 1; i < 8; i++) s += red[row][i];
        rowinv[row] = 1.f / s;
    }
    __syncthreads();

    // coalesced write-out: attn[(b*16+h), m0+rr, c]
    const size_t obase = ((size_t)(b * NHEADS + h) * LSEQ + m0) * LSEQ;
    for (int rr = 0; rr < 32; rr++) {
        const float inv = rowinv[rr];
        for (int c = tid; c < LSEQ; c += 256)
            attn_out[obase + (size_t)rr * LSEQ + c] = strip_s[rr][c] * inv;
    }
}

// ---------------- host launcher ----------------
extern "C" void kernel_launch(void* const* d_in, const int* in_sizes, int n_in,
                              void* d_out, int out_size, void* d_ws, size_t ws_size,
                              hipStream_t stream) {
    const float* x        = (const float*)d_in[0];   // [2,2048,1024]
    const float* W_dq     = (const float*)d_in[1];   // [1024,512]
    const float* W_uq     = (const float*)d_in[2];   // [512,1024]
    const float* q_gamma  = (const float*)d_in[3];   // [512]
    const float* q_beta   = (const float*)d_in[4];   // [512]
    const float* W_dkv    = (const float*)d_in[5];   // [1024,682]
    const float* W_ukv    = (const float*)d_in[6];   // [682,2048]
    const float* kv_gamma = (const float*)d_in[7];   // [682]
    const float* kv_beta  = (const float*)d_in[8];   // [682]

    float* out      = (float*)d_out;
    float* out_x    = out;                                   // 4,194,304
    float* out_attn = out + (size_t)NROWS * D_MODEL;         // 134,217,728
    float* out_ckv  = out_attn + (size_t)BATCH * NHEADS * LSEQ * LSEQ; // 2,793,472

    // workspace carve-out (~52 MB)
    char* ws = (char*)d_ws;
    size_t off = 0;
    auto carve = [&](size_t bytes) { void* p = ws + off; off += (bytes + 255) & ~(size_t)255; return p; };
    _Float16* x_h     = (_Float16*)carve((size_t)NROWS * D_MODEL * 2);   // 8 MB
    _Float16* Wdq_t   = (_Float16*)carve((size_t)QPROJ * D_MODEL * 2);   // 1 MB   [512 x 1024]
    _Float16* Wuq_t   = (_Float16*)carve((size_t)D_MODEL * QPROJ * 2);   // 1 MB   [1024 x 512]
    _Float16* Wdkv_t  = (_Float16*)carve((size_t)KVP_PAD * D_MODEL * 2); // 1.4 MB [704 x 1024]
    _Float16* Wukv_t  = (_Float16*)carve((size_t)D_MODEL * KVP_PAD * 2); // 1.4 MB [1024 x 704]
    float*    raw_buf = (float*)carve((size_t)NROWS * KVP_PAD * 4);      // 11.5 MB (cq_raw then ckv_raw)
    _Float16* cq_h    = (_Float16*)carve((size_t)NROWS * QPROJ * 2);     // 4 MB
    _Float16* Q_h     = (_Float16*)carve((size_t)NROWS * D_MODEL * 2);   // 8 MB
    _Float16* ckv_h   = (_Float16*)carve((size_t)NROWS * KVP_PAD * 2);   // 5.6 MB
    _Float16* K_h     = (_Float16*)carve((size_t)NROWS * D_MODEL * 2);   // 8 MB
    (void)ws_size; (void)in_sizes; (void)n_in; (void)out_size;

    // prep
    convert_x_kernel<<<(NROWS * D_MODEL + 255) / 256, 256, 0, stream>>>(x, out_x, x_h, NROWS * D_MODEL);
    dim3 tb(16, 16);
    transpose_w_kernel<<<dim3(D_MODEL / 16, QPROJ / 16),  tb, 0, stream>>>(W_dq,  Wdq_t,  QPROJ,   D_MODEL, QPROJ,   QPROJ,   D_MODEL);
    transpose_w_kernel<<<dim3(QPROJ / 16,  D_MODEL / 16), tb, 0, stream>>>(W_uq,  Wuq_t,  D_MODEL, QPROJ,   D_MODEL, D_MODEL, QPROJ);
    transpose_w_kernel<<<dim3(D_MODEL / 16, KVP_PAD / 16),tb, 0, stream>>>(W_dkv, Wdkv_t, KVP_PAD, D_MODEL, KVP,     KVP,     D_MODEL);
    transpose_w_kernel<<<dim3(KVP_PAD / 16, D_MODEL / 16),tb, 0, stream>>>(W_ukv, Wukv_t, D_MODEL, KVP,     2 * D_MODEL, 2 * D_MODEL, KVP_PAD);

    // cq_raw = x @ W_dq  (f32)
    gemm_wmma_kernel<<<dim3(QPROJ / 64, NROWS / 128), 256, 0, stream>>>(x_h, Wdq_t, raw_buf, D_MODEL, QPROJ, 0);
    // compressed_q = LN(cq_raw) -> f16
    ln_kernel<<<NROWS, 256, 0, stream>>>(raw_buf, QPROJ, QPROJ, q_gamma, q_beta, nullptr, 0, cq_h, QPROJ, QPROJ);
    // Q = cq @ W_uq  (f16)
    gemm_wmma_kernel<<<dim3(D_MODEL / 64, NROWS / 128), 256, 0, stream>>>(cq_h, Wuq_t, Q_h, QPROJ, D_MODEL, 1);
    // ckv_raw = x @ W_dkv  (f32, padded N=704)
    gemm_wmma_kernel<<<dim3(KVP_PAD / 64, NROWS / 128), 256, 0, stream>>>(x_h, Wdkv_t, raw_buf, D_MODEL, KVP_PAD, 0);
    // compressed_kv = LN(ckv_raw) -> output #3 (f32) + f16 padded
    ln_kernel<<<NROWS, 256, 0, stream>>>(raw_buf, KVP_PAD, KVP, kv_gamma, kv_beta, out_ckv, KVP, ckv_h, KVP_PAD, KVP_PAD);
    // K = ckv @ W_ukv[:, :1024]  (f16; V half is dead code in the reference outputs)
    gemm_wmma_kernel<<<dim3(D_MODEL / 64, NROWS / 128), 256, 0, stream>>>(ckv_h, Wukv_t, K_h, KVP_PAD, D_MODEL, 1);
    // attn = softmax(Q K^T / 8) per (b,h) -> output #2
    attn_kernel<<<dim3(LSEQ / 32, NHEADS, BATCH), 256, 0, stream>>>(Q_h, K_h, out_attn);
}
